// GCNLayer_64871186039446
// MI455X (gfx1250) — compile-verified, matrix operand order checked by reading, not run on hardware
//
#include <hip/hip_runtime.h>

#define NN 256    // nodes
#define NT 2048   // times
#define NF 128    // features

typedef __attribute__((ext_vector_type(16))) __bf16 bf16x16;
typedef __attribute__((ext_vector_type(8)))  __bf16 bf16x8;
typedef __attribute__((ext_vector_type(2)))  __bf16 bf16x2;
typedef __attribute__((ext_vector_type(8)))  float  v8f;
typedef __attribute__((ext_vector_type(4)))  float  f32x4;
typedef __attribute__((ext_vector_type(4)))  unsigned int u32x4;

union FragU { bf16x16 v; bf16x2 p[8]; u32x4 q[2]; };
union HalfU { bf16x8  v; bf16x2 p[4]; u32x4 q;    };

// f32 -> bf16: native fptrunc (RNE) -> v_cvt_pk_bf16_f32 on gfx1250 (verified in disasm).
__device__ __forceinline__ __bf16 f2bf(float f){ return (__bf16)f; }

__device__ __forceinline__ bf16x2 pk2(float a, float b){
#if __has_builtin(__builtin_amdgcn_cvt_pk_bf16_f32)
  return __builtin_amdgcn_cvt_pk_bf16_f32(a, b);
#else
  bf16x2 r; r[0] = (__bf16)a; r[1] = (__bf16)b; return r;
#endif
}

// Load a 16x32 bf16 fragment (A layout; B loaded as rows of B^T with same layout).
// lane m (0-15): row m, K chunks [0..7] and [16..23]
// lane m (16-31): row m-16, K chunks [8..15] and [24..31]
__device__ __forceinline__ bf16x16 load_frag_bf16(const __bf16* tile, int stride, int lane){
  const __bf16* r0 = tile + (size_t)(lane & 15) * stride + ((lane >> 4) << 3);
  FragU f;
  f.q[0] = *(const u32x4*)(r0);
  f.q[1] = *(const u32x4*)(r0 + 16);
  return f.v;
}

// Same fragment, sourced from f32 memory with on-the-fly bf16 conversion.
__device__ __forceinline__ bf16x16 load_frag_f32(const float* tile, size_t stride, int lane){
  const float* r0 = tile + (size_t)(lane & 15) * stride + ((lane >> 4) << 3);
  f32x4 a0 = *(const f32x4*)(r0);
  f32x4 a1 = *(const f32x4*)(r0 + 4);
  f32x4 a2 = *(const f32x4*)(r0 + 16);
  f32x4 a3 = *(const f32x4*)(r0 + 20);
  FragU f;
  f.p[0] = pk2(a0[0], a0[1]);  f.p[1] = pk2(a0[2], a0[3]);
  f.p[2] = pk2(a1[0], a1[1]);  f.p[3] = pk2(a1[2], a1[3]);
  f.p[4] = pk2(a2[0], a2[1]);  f.p[5] = pk2(a2[2], a2[3]);
  f.p[6] = pk2(a3[0], a3[1]);  f.p[7] = pk2(a3[2], a3[3]);
  return f.v;
}

#define WMMA_BF16(a,b,c) \
  __builtin_amdgcn_wmma_f32_16x16x32_bf16(false,(a),false,(b),(short)0,(c),false,false)

#define XW_STRIDE 264   // 256 + 8 bf16 pad -> 132-dword row stride, conflict-free b128 LDS access

// ---- prep: adj_hat = rownorm(adj + I) -> bf16 ; Wt = W^T -> bf16 ----
__global__ __launch_bounds__(256) void gcn_prep(const float* __restrict__ adj,
                                                const float* __restrict__ W,
                                                __bf16* __restrict__ adjh,
                                                __bf16* __restrict__ Wt){
  if (blockIdx.x < NN) {
    __shared__ float red[NN];
    const int m = blockIdx.x, n = threadIdx.x;
    float v = adj[m * NN + n] + ((m == n) ? 1.0f : 0.0f);
    red[n] = v;
    __syncthreads();
    for (int s = NN / 2; s > 0; s >>= 1) {
      if (n < s) red[n] += red[n + s];
      __syncthreads();
    }
    adjh[m * NN + n] = f2bf(v / red[0]);
  } else {
    const int idx = (blockIdx.x - NN) * 256 + threadIdx.x;   // 0..NF*NF-1
    const int a = idx >> 7, o = idx & (NF - 1);
    Wt[o * NF + a] = f2bf(W[a * NF + o]);
  }
}

// ---- fused: per time-step t, out[:,t,:] = adjh @ (X[:,t,:] @ W) + bias ----
// launch_bounds(256, 4): 4 waves/SIMD resident. Outer loops are deliberately
// rolled (#pragma unroll 1) so LICM cannot hoist all 32 Wt fragments into
// VGPRs -- that hoisting is what forced scratch spills under the register cap.
// Reloads hit WGP$/LDS, which is cheap; occupancy hides the strided X loads.
__global__ __launch_bounds__(256, 4) void gcn_fused(const float* __restrict__ X,
                                                    const __bf16* __restrict__ adjh,
                                                    const __bf16* __restrict__ Wt,
                                                    const float* __restrict__ bias,
                                                    float* __restrict__ out){
  __shared__ __bf16 XWt[NF * XW_STRIDE];   // xw_t stored transposed [o][node], 67,584 B

  const int t    = blockIdx.x;
  const int lane = threadIdx.x & 31;
  const int w    = threadIdx.x >> 5;       // 8 waves
  const int half = lane >> 4;
  const int r16  = lane & 15;

  // ---------- Stage 1: XWt[o][n] = sum_a X[n,t,a] * W[a,o]  (bf16 in LDS) ----------
  // Wave w owns node-tiles {2w, 2w+1} across ALL o-tiles: each X element is
  // loaded+converted exactly once per WG.
#pragma unroll 1
  for (int ni = 0; ni < 2; ++ni) {
    const int nt = w * 2 + ni;
    const float* xb = X + ((size_t)(nt * 16) * NT + t) * NF;   // node-row stride NT*NF
    bf16x16 af[4];
#pragma unroll
    for (int k = 0; k < 4; ++k)
      af[k] = load_frag_f32(xb + k * 32, (size_t)NT * NF, lane);

#pragma unroll 1
    for (int ot = 0; ot < 8; ++ot) {
      v8f acc = {};
#pragma unroll
      for (int k = 0; k < 4; ++k) {
        bf16x16 bf = load_frag_bf16(Wt + (size_t)(ot * 16) * NF + k * 32, NF, lane);
        acc = WMMA_BF16(af[k], bf, acc);
      }
      // C layout: VGPR r -> (node = nt*16 + half*8 + r, o = ot*16 + r16); store transposed
      HalfU h;
      h.p[0] = pk2(acc[0], acc[1]);  h.p[1] = pk2(acc[2], acc[3]);
      h.p[2] = pk2(acc[4], acc[5]);  h.p[3] = pk2(acc[6], acc[7]);
      *(u32x4*)&XWt[(size_t)(ot * 16 + r16) * XW_STRIDE + nt * 16 + half * 8] = h.q;
    }
  }
  __syncthreads();

  // ---------- Stage 2: out[m,t,o] = sum_n adjh[m,n] * XWt[o][n] + bias[o] ----------
  // wave w owns m-tiles {2w, 2w+1}; all 8 o-tiles accumulated per m-tile
  float bo[8];
#pragma unroll
  for (int ot = 0; ot < 8; ++ot) bo[ot] = bias[ot * 16 + r16];

#pragma unroll 1
  for (int mi = 0; mi < 2; ++mi) {
    const int mt = w * 2 + mi;
    v8f acc[8] = {};
    const __bf16* arow = adjh + (size_t)(mt * 16) * NN;
#pragma unroll 1
    for (int k = 0; k < 8; ++k) {                     // K = 256 nodes, 32 per step
      bf16x16 af = load_frag_bf16(arow + k * 32, NN, lane);
#pragma unroll
      for (int ot = 0; ot < 8; ++ot) {
        bf16x16 bf = load_frag_bf16(&XWt[(size_t)(ot * 16) * XW_STRIDE + k * 32],
                                    XW_STRIDE, lane);
        acc[ot] = WMMA_BF16(af, bf, acc[ot]);
      }
    }
#pragma unroll
    for (int ot = 0; ot < 8; ++ot) {
      const int oc = ot * 16 + r16;
      const size_t base = ((size_t)(mt * 16 + half * 8) * NT + t) * NF + oc;
#pragma unroll
      for (int r = 0; r < 8; ++r)
        out[base + (size_t)r * NT * NF] = acc[ot][r] + bo[ot];
    }
  }
}

extern "C" void kernel_launch(void* const* d_in, const int* in_sizes, int n_in,
                              void* d_out, int out_size, void* d_ws, size_t ws_size,
                              hipStream_t stream) {
  (void)in_sizes; (void)n_in; (void)out_size; (void)ws_size;
  const float* X    = (const float*)d_in[0];   // [256,2048,128] f32
  const float* adj  = (const float*)d_in[1];   // [256,256] f32
  const float* W    = (const float*)d_in[2];   // [128,128] f32
  const float* bias = (const float*)d_in[3];   // [128] f32

  __bf16* adjh = (__bf16*)d_ws;                // 256*256 bf16 = 128 KB
  __bf16* Wt   = adjh + (size_t)NN * NN;       // 128*128 bf16 =  32 KB

  gcn_prep<<<NN + (NF * NF) / 256, 256, 0, stream>>>(adj, W, adjh, Wt);
  gcn_fused<<<NT, 256, 0, stream>>>(X, adjh, Wt, bias, (float*)d_out);
}
